// BatchMultiHeadGraphAttention_12618613916163
// MI455X (gfx1250) — matrix operand order repaired and checked by reading, not compiled, not run on hardware
//
#include <hip/hip_runtime.h>
#include <hip/hip_bf16.h>

#define BSZ 4
#define NN 2048
#define NH 2
#define FIN 768
#define FOUT 768
#define NEG_SLOPE 0.2f

#define LDA  40    // 128x32 bf16 A tile: padded stride (80B rows, 16B-aligned chunks)
#define LDB2 144   // 32x128 bf16 B tile: padded stride (288B rows, 32B-aligned frags)

typedef __attribute__((ext_vector_type(16))) __bf16 v16bf;
typedef __attribute__((ext_vector_type(8)))  __bf16 v8bf;
typedef __attribute__((ext_vector_type(2)))  __bf16 v2bf;
typedef __attribute__((ext_vector_type(8)))  float  v8f;
typedef __attribute__((ext_vector_type(2)))  float  v2f;

// native f32 -> bf16 conversions (lower to v_cvt_pk_bf16_f32)
__device__ __forceinline__ unsigned short f2bf(float x) {
    union { __bf16 b; unsigned short u; } c;
    c.b = (__bf16)x;
    return c.u;
}
__device__ __forceinline__ float bf2f(unsigned short u) {
    return __uint_as_float(((unsigned int)u) << 16);
}
__device__ __forceinline__ unsigned int pk2(float x, float y) {
    v2f f = {x, y};
    union { v2bf b; unsigned int u; } c;
    c.b = __builtin_convertvector(f, v2bf);
    return c.u;
}

// ---------------------------------------------------------------------------
// Kernel 1: h_prime[bh, m, o] = sum_k h[b,m,k] * w[hd,k,o]   (bf16 out)
// Block tile 128(M) x 128(N), 8 waves (4x2), each wave 2x4 frags of 16x16.
// Software-pipelined: tile k+1 global f32 loads overlap tile-k WMMAs.
// ---------------------------------------------------------------------------
__global__ __launch_bounds__(256)
void hprime_gemm_kernel(const float* __restrict__ h, const float* __restrict__ w,
                        unsigned short* __restrict__ hp)
{
    __shared__ __align__(32) unsigned short As[128 * LDA];
    __shared__ __align__(32) unsigned short Bs[32 * LDB2];

    const int tid  = threadIdx.x;
    const int lane = tid & 31, wave = tid >> 5;
    const int wm = wave & 3, wn = wave >> 2;
    const int n0 = blockIdx.x * 128;
    const int m0 = blockIdx.y * 128;
    const int bh = blockIdx.z;
    const int b = bh >> 1, hd = bh & 1;

    v8f acc[2][4];
    #pragma unroll
    for (int i = 0; i < 2; i++)
        #pragma unroll
        for (int j = 0; j < 4; j++)
            #pragma unroll
            for (int e = 0; e < 8; e++) acc[i][j][e] = 0.f;

    const int c0   = (lane & 16) ? 8 : 0;          // A-frag K chunk base per ISA layout
    const int arow = wm * 32 + (lane & 15);

    float4 ra[4], rb[4];
    auto load_tile = [&](int k0) {
        #pragma unroll
        for (int i = 0; i < 4; i++) {
            int idx = tid + 256 * i;
            int r = idx >> 3, c4 = idx & 7;
            ra[i] = *(const float4*)&h[(size_t)(b * NN + m0 + r) * FIN + k0 + c4 * 4];
        }
        #pragma unroll
        for (int i = 0; i < 4; i++) {
            int idx = tid + 256 * i;
            int r = idx >> 5, c4 = idx & 31;
            rb[i] = *(const float4*)&w[(size_t)(hd * FIN + k0 + r) * FOUT + n0 + c4 * 4];
        }
    };

    load_tile(0);   // prologue

    for (int k0 = 0; k0 < FIN; k0 += 32) {
        // convert + store current tile (registers -> LDS bf16)
        #pragma unroll
        for (int i = 0; i < 4; i++) {
            int idx = tid + 256 * i;
            int r = idx >> 3, c4 = idx & 7;
            uint2 pk; pk.x = pk2(ra[i].x, ra[i].y); pk.y = pk2(ra[i].z, ra[i].w);
            *(uint2*)&As[r * LDA + c4 * 4] = pk;
        }
        #pragma unroll
        for (int i = 0; i < 4; i++) {
            int idx = tid + 256 * i;
            int r = idx >> 5, c4 = idx & 31;
            uint2 pk; pk.x = pk2(rb[i].x, rb[i].y); pk.y = pk2(rb[i].z, rb[i].w);
            *(uint2*)&Bs[r * LDB2 + c4 * 4] = pk;
        }
        __syncthreads();

        if (k0 + 32 < FIN) load_tile(k0 + 32);   // overlap next loads with WMMAs

        v16bf af[2], bfr[4];
        #pragma unroll
        for (int fi = 0; fi < 2; fi++) {
            const unsigned short* ap = &As[(arow + fi * 16) * LDA + c0];
            union { v16bf v; v8bf h2[2]; } u;
            u.h2[0] = *(const v8bf*)ap;
            u.h2[1] = *(const v8bf*)(ap + 16);
            af[fi] = u.v;
        }
        #pragma unroll
        for (int fj = 0; fj < 4; fj++)
        {
            union { v16bf v; unsigned short u[16]; } t;
            #pragma unroll
            for (int s = 0; s < 16; s++)
                t.u[s] = Bs[(c0 + s + ((s >= 8) ? 8 : 0)) * LDB2 + wn * 64 + fj * 16 + (lane & 15)];
            bfr[fj] = t.v;
        }

        #pragma unroll
        for (int fi = 0; fi < 2; fi++)
            #pragma unroll
            for (int fj = 0; fj < 4; fj++)
                acc[fi][fj] = __builtin_amdgcn_wmma_f32_16x16x32_bf16(
                    false, af[fi], false, bfr[fj], (short)0, acc[fi][fj], false, false);
        __syncthreads();
    }

    const int mbase = m0 + wm * 32 + ((lane & 16) ? 8 : 0);
    const int nbase = n0 + wn * 64 + (lane & 15);
    #pragma unroll
    for (int fi = 0; fi < 2; fi++)
        #pragma unroll
        for (int fj = 0; fj < 4; fj++)
            #pragma unroll
            for (int v = 0; v < 8; v++) {
                int m = mbase + fi * 16 + v;
                int n = nbase + fj * 16;
                hp[(size_t)(bh * NN + m) * FOUT + n] = f2bf(acc[fi][fj][v]);
            }
}

// ---------------------------------------------------------------------------
// Kernel 2: attn_src/dst[bh, i] = sum_o tanh(hp[bh,i,o]) * a_{src,dst}[hd,o]
// ---------------------------------------------------------------------------
__global__ __launch_bounds__(256)
void attn_vec_kernel(const unsigned short* __restrict__ hp,
                     const float* __restrict__ a_src, const float* __restrict__ a_dst,
                     float* __restrict__ vsrc, float* __restrict__ vdst)
{
    __shared__ float r1[256];
    __shared__ float r2[256];
    const int i = blockIdx.x, bh = blockIdx.y;
    const int hd = bh & 1;
    const int tid = threadIdx.x;
    const unsigned short* row = &hp[(size_t)(bh * NN + i) * FOUT];
    float s1 = 0.f, s2 = 0.f;
    for (int o = tid; o < FOUT; o += 256) {
        float t = tanhf(bf2f(row[o]));
        s1 += t * a_src[hd * FOUT + o];
        s2 += t * a_dst[hd * FOUT + o];
    }
    r1[tid] = s1; r2[tid] = s2; __syncthreads();
    for (int off = 128; off > 0; off >>= 1) {
        if (tid < off) { r1[tid] += r1[tid + off]; r2[tid] += r2[tid + off]; }
        __syncthreads();
    }
    if (tid == 0) { vsrc[bh * NN + i] = r1[0]; vdst[bh * NN + i] = r2[0]; }
}

// ---------------------------------------------------------------------------
// Kernel 3: masked softmax row -> P[bh, i, j] in bf16
// keep = adj[b,i,j] || (i==j)   (diagonal always kept -> row sum > 0)
// ---------------------------------------------------------------------------
__global__ __launch_bounds__(256)
void softmax_p_kernel(const unsigned char* __restrict__ adj,
                      const float* __restrict__ vsrc, const float* __restrict__ vdst,
                      unsigned short* __restrict__ P)
{
    __shared__ float red[256];
    const int i = blockIdx.x, bh = blockIdx.y;
    const int b = bh >> 1;
    const int tid = threadIdx.x;
    const unsigned char* arow = &adj[(size_t)(b * NN + i) * NN];
    const float* drow = &vdst[bh * NN];
    const float si = vsrc[bh * NN + i];

    float mx = -INFINITY;
    for (int j = tid; j < NN; j += 256) {
        if (arow[j] || j == i) {
            float sc = si + drow[j];
            sc = (sc >= 0.f) ? sc : NEG_SLOPE * sc;
            mx = fmaxf(mx, sc);
        }
    }
    red[tid] = mx; __syncthreads();
    for (int off = 128; off > 0; off >>= 1) {
        if (tid < off) red[tid] = fmaxf(red[tid], red[tid + off]);
        __syncthreads();
    }
    mx = red[0]; __syncthreads();

    float sm = 0.f;
    for (int j = tid; j < NN; j += 256) {
        if (arow[j] || j == i) {
            float sc = si + drow[j];
            sc = (sc >= 0.f) ? sc : NEG_SLOPE * sc;
            sm += __expf(sc - mx);
        }
    }
    red[tid] = sm; __syncthreads();
    for (int off = 128; off > 0; off >>= 1) {
        if (tid < off) red[tid] += red[tid + off];
        __syncthreads();
    }
    const float sinv = 1.f / red[0];

    unsigned short* prow = &P[(size_t)(bh * NN + i) * NN];
    for (int j = tid; j < NN; j += 256) {
        float p = 0.f;
        if (arow[j] || j == i) {
            float sc = si + drow[j];
            sc = (sc >= 0.f) ? sc : NEG_SLOPE * sc;
            p = __expf(sc - mx) * sinv;
        }
        prow[j] = f2bf(p);
    }
}

// ---------------------------------------------------------------------------
// Kernel 4: out[bh, m, o] = sum_j P[bh,m,j] * hp[bh,j,o] + bias[o]
// 128x128 block tile; double-buffered LDS fed by CDNA5 async Global->LDS
// copies (global_load_async_to_lds_b128, tracked by ASYNCcnt).
// ---------------------------------------------------------------------------
__device__ __forceinline__ void async_b128(unsigned lds_off, const void* gptr)
{
    asm volatile("global_load_async_to_lds_b128 %0, %1, off"
                 :: "v"(lds_off), "v"((unsigned long long)(uintptr_t)gptr)
                 : "memory");
}

__global__ __launch_bounds__(256)
void out_gemm_kernel(const unsigned short* __restrict__ P,
                     const unsigned short* __restrict__ hp,
                     const float* __restrict__ bias,
                     float* __restrict__ out)
{
    __shared__ __align__(32) unsigned short As[2][128 * LDA];
    __shared__ __align__(32) unsigned short Bs[2][32 * LDB2];

    const int tid  = threadIdx.x;
    const int lane = tid & 31, wave = tid >> 5;
    const int wm = wave & 3, wn = wave >> 2;
    const int n0 = blockIdx.x * 128;
    const int m0 = blockIdx.y * 128;
    const int bh = blockIdx.z;

    v8f acc[2][4];
    #pragma unroll
    for (int i = 0; i < 2; i++)
        #pragma unroll
        for (int j = 0; j < 4; j++)
            #pragma unroll
            for (int e = 0; e < 8; e++) acc[i][j][e] = 0.f;

    const int c0   = (lane & 16) ? 8 : 0;
    const int arow = wm * 32 + (lane & 15);

    // this thread's two 16B chunks of the A tile and two of the B tile
    const int ar0 = (tid      ) >> 2, ac0 = (tid      ) & 3;
    const int ar1 = (tid + 256) >> 2, ac1 = (tid + 256) & 3;
    const int br0 = (tid      ) >> 4, bc0 = (tid      ) & 15;
    const int br1 = (tid + 256) >> 4, bc1 = (tid + 256) & 15;

    const unsigned short* Pbase  = &P[(size_t)(bh * NN + m0) * NN];
    const unsigned short* HPbase = &hp[(size_t)bh * NN * FOUT + n0];

    auto issue_tile = [&](int k0, int buf) {
        async_b128((unsigned)(uintptr_t)&As[buf][ar0 * LDA + ac0 * 8],
                   Pbase + (size_t)ar0 * NN + k0 + ac0 * 8);
        async_b128((unsigned)(uintptr_t)&As[buf][ar1 * LDA + ac1 * 8],
                   Pbase + (size_t)ar1 * NN + k0 + ac1 * 8);
        async_b128((unsigned)(uintptr_t)&Bs[buf][br0 * LDB2 + bc0 * 8],
                   HPbase + (size_t)(k0 + br0) * FOUT + bc0 * 8);
        async_b128((unsigned)(uintptr_t)&Bs[buf][br1 * LDB2 + bc1 * 8],
                   HPbase + (size_t)(k0 + br1) * FOUT + bc1 * 8);
    };

    issue_tile(0, 0);   // prologue

    for (int k0 = 0; k0 < NN; k0 += 32) {
        const int buf = (k0 >> 5) & 1;
        // my async copies for `buf` done; barrier makes everyone's visible
        asm volatile("s_wait_asynccnt 0x0" ::: "memory");
        __syncthreads();
        if (k0 + 32 < NN) issue_tile(k0 + 32, buf ^ 1);   // overlap DMA with math

        v16bf af[2], bfr[4];
        #pragma unroll
        for (int fi = 0; fi < 2; fi++) {
            const unsigned short* ap = &As[buf][(arow + fi * 16) * LDA + c0];
            union { v16bf v; v8bf h2[2]; } u;
            u.h2[0] = *(const v8bf*)ap;
            u.h2[1] = *(const v8bf*)(ap + 16);
            af[fi] = u.v;
        }
        #pragma unroll
        for (int fj = 0; fj < 4; fj++)
        {
            union { v16bf v; unsigned short u[16]; } t;
            #pragma unroll
            for (int s = 0; s < 16; s++)
                t.u[s] = Bs[buf][(c0 + s + ((s >= 8) ? 8 : 0)) * LDB2 + wn * 64 + fj * 16 + (lane & 15)];
            bfr[fj] = t.v;
        }

        #pragma unroll
        for (int fi = 0; fi < 2; fi++)
            #pragma unroll
            for (int fj = 0; fj < 4; fj++)
                acc[fi][fj] = __builtin_amdgcn_wmma_f32_16x16x32_bf16(
                    false, af[fi], false, bfr[fj], (short)0, acc[fi][fj], false, false);
    }

    const int mbase = m0 + wm * 32 + ((lane & 16) ? 8 : 0);
    const int nbase = n0 + wn * 64 + (lane & 15);
    #pragma unroll
    for (int fi = 0; fi < 2; fi++)
        #pragma unroll
        for (int fj = 0; fj < 4; fj++)
            #pragma unroll
            for (int v = 0; v < 8; v++) {
                int m = mbase + fi * 16 + v;
                int n = nbase + fj * 16;
                out[(size_t)(bh * NN + m) * FOUT + n] = acc[fi][fj][v] + bias[n];
            }
}

// ---------------------------------------------------------------------------
extern "C" void kernel_launch(void* const* d_in, const int* in_sizes, int n_in,
                              void* d_out, int out_size, void* d_ws, size_t ws_size,
                              hipStream_t stream)
{
    const float*         h     = (const float*)d_in[0];
    const unsigned char* adj   = (const unsigned char*)d_in[1];   // jax bool, 1 byte
    const float*         w     = (const float*)d_in[2];
    const float*         a_src = (const float*)d_in[3];
    const float*         a_dst = (const float*)d_in[4];
    const float*         bias  = (const float*)d_in[5];
    float*               out   = (float*)d_out;

    // workspace layout (needs ~93 MB):
    //   hp  bf16 [8][2048][768]   = 25,165,824 B
    //   P   bf16 [8][2048][2048]  = 67,108,864 B
    //   vsrc/vdst f32 [8*2048]    =   131,072 B
    char* ws = (char*)d_ws;
    const size_t HP_BYTES = (size_t)8 * NN * FOUT * 2;
    const size_t P_BYTES  = (size_t)8 * NN * NN * 2;
    unsigned short* hp   = (unsigned short*)ws;
    unsigned short* P    = (unsigned short*)(ws + HP_BYTES);
    float*          vsrc = (float*)(ws + HP_BYTES + P_BYTES);
    float*          vdst = vsrc + 8 * NN;

    hprime_gemm_kernel<<<dim3(FOUT / 128, NN / 128, 8), 256, 0, stream>>>(h, w, hp);
    attn_vec_kernel<<<dim3(NN, 8), 256, 0, stream>>>(hp, a_src, a_dst, vsrc, vdst);
    softmax_p_kernel<<<dim3(NN, 8), 256, 0, stream>>>(adj, vsrc, vdst, P);
    out_gemm_kernel<<<dim3(FOUT / 128, NN / 128, 8), 256, 0, stream>>>(P, hp, bias, out);
}
